// ContrastiveLoss_7172595385025
// MI455X (gfx1250) — matrix operand order repaired and checked by reading, not compiled, not run on hardware
//
#include <hip/hip_runtime.h>

// ContrastiveLoss on MI455X (gfx1250): Gram matrix via split-f16 WMMA
// (hi*hi + hi*lo + lo*hi in f32 accumulators ~ f32 precision), fused
// per-pair loss epilogue, deterministic two-level reduction.

#define B_ROWS 8192
#define D_COLS 256
#define NT     64            // 8192 / 128 tiles per dimension

static __device__ __constant__ float kMargin = 1.0f;
static __device__ __constant__ float kEps    = 1e-6f;

typedef __attribute__((ext_vector_type(16))) _Float16 v16h;
typedef __attribute__((ext_vector_type(8)))  _Float16 v8h;
typedef __attribute__((ext_vector_type(8)))  float    v8f;

// ---------------------------------------------------------------------------
// Phase 0: f32 -> (hi, lo) f16 split + per-row sum / squared-norm.
// One 256-thread block per row (D == 256). Deterministic LDS tree reduce.
// ---------------------------------------------------------------------------
__global__ __launch_bounds__(256) void prep_kernel(
    const float* __restrict__ src,
    _Float16* __restrict__ hi, _Float16* __restrict__ lo,
    float* __restrict__ sq_norm, float* __restrict__ row_sum) {
  const int row = blockIdx.x;
  const int tid = threadIdx.x;
  const float x = src[(size_t)row * D_COLS + tid];

  const _Float16 h = (_Float16)x;
  const _Float16 l = (_Float16)(x - (float)h);
  hi[(size_t)row * D_COLS + tid] = h;
  lo[(size_t)row * D_COLS + tid] = l;

  __shared__ float s1[256];
  __shared__ float s2[256];
  s1[tid] = x;
  s2[tid] = x * x;
  __syncthreads();
#pragma unroll
  for (int off = 128; off > 0; off >>= 1) {
    if (tid < off) {
      s1[tid] += s1[tid + off];
      s2[tid] += s2[tid + off];
    }
    __syncthreads();
  }
  if (tid == 0) {
    row_sum[row] = s1[0];
    sq_norm[row] = s2[0];
  }
}

// ---------------------------------------------------------------------------
// Fragment loaders (wave32 WMMA 16-bit layouts, CDNA5 ISA 7.12.2).
// A 16x32: lanes 0-15 hold K = {0..7, 16..23}; lanes 16-31 K = {8..15, 24..31}.
// B 32x16: same K split as A (lanes 0-15 K = {0..7, 16..23}; lanes 16-31 K = {8..15, 24..31}).
// row/col argument already includes the laneHalf (M / N) component.
// ---------------------------------------------------------------------------
__device__ __forceinline__ v16h load_a_frag(const _Float16* __restrict__ base,
                                            int row, int k0, int laneHi) {
  const _Float16* p = base + (size_t)row * D_COLS + k0 + laneHi * 8;
  union { v16h v; v8h h[2]; } f;
  f.h[0] = *(const v8h*)(p);
  f.h[1] = *(const v8h*)(p + 16);
  return f.v;
}

__device__ __forceinline__ v16h load_b_frag(const _Float16* __restrict__ base,
                                            int col, int k0, int laneHi) {
  const _Float16* p = base + (size_t)col * D_COLS + k0 + laneHi * 8;
  union { v16h v; v8h h[2]; } f;
  f.h[0] = *(const v8h*)(p);
  f.h[1] = *(const v8h*)(p + 16);
  return f.v;
}

// ---------------------------------------------------------------------------
// Phase 1: 128x128 pair tile per block; 8 waves as 2x4; each wave owns a
// 64x32 region = 4x2 tiles of 16x16 f32 accumulators. K loop: 8 steps of 32,
// 3 WMMAs (hi*hi, hi*lo, lo*hi) per tile per step. Epilogue fuses the
// contrastive-loss math with the strict upper-triangle mask j > i.
// ---------------------------------------------------------------------------
__global__ __launch_bounds__(256) void gemm_loss_kernel(
    const _Float16* __restrict__ Ahi, const _Float16* __restrict__ Alo,
    const float* __restrict__ sq_norm, const float* __restrict__ row_sum,
    const int* __restrict__ label, float* __restrict__ partial) {
  const int tj  = blockIdx.x;           // column tile
  const int ti  = blockIdx.y;           // row tile
  const int bid = ti * gridDim.x + tj;

  if (tj < ti) {                        // strictly-lower tile: contributes 0
    if (threadIdx.x == 0) partial[bid] = 0.0f;
    return;
  }

  const int lane     = threadIdx.x & 31;
  const int wave     = threadIdx.x >> 5;
  const int wm       = wave >> 2;       // 0..1
  const int wn       = wave & 3;        // 0..3
  const int laneHalf = lane & 15;
  const int laneHi   = lane >> 4;

  const int rowBase = ti * 128 + wm * 64;
  const int colBase = tj * 128 + wn * 32;

  v8f acc[4][2];
#pragma unroll
  for (int tm = 0; tm < 4; ++tm)
#pragma unroll
    for (int tn = 0; tn < 2; ++tn) acc[tm][tn] = (v8f){};

#pragma unroll 1
  for (int k0 = 0; k0 < D_COLS; k0 += 32) {
    v16h ahi[4], alo[4], bhi[2], blo[2];
#pragma unroll
    for (int tm = 0; tm < 4; ++tm) {
      const int r = rowBase + tm * 16 + laneHalf;
      ahi[tm] = load_a_frag(Ahi, r, k0, laneHi);
      alo[tm] = load_a_frag(Alo, r, k0, laneHi);
    }
#pragma unroll
    for (int tn = 0; tn < 2; ++tn) {
      const int c = colBase + tn * 16 + laneHalf;
      bhi[tn] = load_b_frag(Ahi, c, k0, laneHi);
      blo[tn] = load_b_frag(Alo, c, k0, laneHi);
    }
#pragma unroll
    for (int tm = 0; tm < 4; ++tm) {
#pragma unroll
      for (int tn = 0; tn < 2; ++tn) {
        v8f c = acc[tm][tn];
        c = __builtin_amdgcn_wmma_f32_16x16x32_f16(false, ahi[tm], false, bhi[tn],
                                                   (short)0, c, false, false);
        c = __builtin_amdgcn_wmma_f32_16x16x32_f16(false, ahi[tm], false, blo[tn],
                                                   (short)0, c, false, false);
        c = __builtin_amdgcn_wmma_f32_16x16x32_f16(false, alo[tm], false, bhi[tn],
                                                   (short)0, c, false, false);
        acc[tm][tn] = c;
      }
    }
  }

  // Fused epilogue. C layout: lanes 0-15 -> N = laneHalf, M = e;
  // lanes 16-31 -> N = laneHalf, M = e + 8 (e = accumulator VGPR index).
  const float dEps2 = (float)D_COLS * kEps * kEps;
  float lsum = 0.0f;
#pragma unroll
  for (int tn = 0; tn < 2; ++tn) {
    const int   j     = colBase + tn * 16 + laneHalf;
    const float sqn_j = sq_norm[j];
    const float rs_j  = row_sum[j];
    const int   lb_j  = label[j];
#pragma unroll
    for (int tm = 0; tm < 4; ++tm) {
      const int ibase = rowBase + tm * 16 + laneHi * 8;
#pragma unroll
      for (int e = 0; e < 8; ++e) {
        const int i = ibase + e;
        if (j > i) {
          const float g      = acc[tm][tn][e];
          const float sq     = sq_norm[i] + sqn_j - 2.0f * g;
          const float sq_eps = sq + 2.0f * kEps * (row_sum[i] - rs_j) + dEps2;
          const float se     = fmaxf(sq_eps, 1e-12f);
          float val;
          if (label[i] == lb_j) {
            val = se;                                  // dist^2
          } else {
            const float m = fmaxf(kMargin - sqrtf(se), 0.0f);
            val = m * m;
          }
          lsum += val;
        }
      }
    }
  }

  __shared__ float red[256];
  red[threadIdx.x] = lsum;
  __syncthreads();
#pragma unroll
  for (int off = 128; off > 0; off >>= 1) {
    if (threadIdx.x < off) red[threadIdx.x] += red[threadIdx.x + off];
    __syncthreads();
  }
  if (threadIdx.x == 0) partial[bid] = red[0];
}

// ---------------------------------------------------------------------------
// Phase 2: deterministic final reduction of per-block partials.
// ---------------------------------------------------------------------------
__global__ __launch_bounds__(256) void reduce_kernel(
    const float* __restrict__ partial, int n,
    float* __restrict__ out, float inv_cnt) {
  __shared__ float red[256];
  float s = 0.0f;
  for (int t = threadIdx.x; t < n; t += 256) s += partial[t];
  red[threadIdx.x] = s;
  __syncthreads();
#pragma unroll
  for (int off = 128; off > 0; off >>= 1) {
    if (threadIdx.x < off) red[threadIdx.x] += red[threadIdx.x + off];
    __syncthreads();
  }
  if (threadIdx.x == 0) out[0] = red[0] * inv_cnt;
}

// ---------------------------------------------------------------------------
extern "C" void kernel_launch(void* const* d_in, const int* in_sizes, int n_in,
                              void* d_out, int out_size, void* d_ws, size_t ws_size,
                              hipStream_t stream) {
  (void)in_sizes; (void)n_in; (void)out_size; (void)ws_size;

  const float* src   = (const float*)d_in[0];
  const int*   label = (const int*)d_in[1];

  // Workspace layout (total ~8.4 MB):
  //   [0, 4MB)        hi   f16 [B][D]
  //   [4MB, 8MB)      lo   f16 [B][D]
  //   then sq_norm[B], row_sum[B], partial[NT*NT] floats
  char* ws = (char*)d_ws;
  const size_t matBytes = (size_t)B_ROWS * D_COLS * sizeof(_Float16);
  _Float16* hi      = (_Float16*)(ws);
  _Float16* lo      = (_Float16*)(ws + matBytes);
  float*    sq_norm = (float*)(ws + 2 * matBytes);
  float*    row_sum = sq_norm + B_ROWS;
  float*    partial = row_sum + B_ROWS;

  prep_kernel<<<B_ROWS, 256, 0, stream>>>(src, hi, lo, sq_norm, row_sum);

  dim3 grid(NT, NT);
  gemm_loss_kernel<<<grid, 256, 0, stream>>>(hi, lo, sq_norm, row_sum, label,
                                             partial);

  const double cnt     = (double)B_ROWS * (double)(B_ROWS - 1) * 0.5;
  const float  inv_cnt = (float)(1.0 / (cnt + 1e-6));
  reduce_kernel<<<1, 256, 0, stream>>>(partial, NT * NT, (float*)d_out, inv_cnt);
}
